// MHSAttention_21337397527391
// MI455X (gfx1250) — matrix operand order repaired and checked by reading, not compiled, not run on hardware
//
#include <hip/hip_runtime.h>
#include <hip/hip_bf16.h>

// ---------------------------------------------------------------------------
// MHSA with 2D rotary, MI455X (gfx1250, wave32, WMMA bf16 -> f32 accumulate)
//  - GEMMs: one wave = 32x64 output tile (2 A-frags x 4 B-frags -> 8 WMMA per
//    k-step; 12 b128 loads per 8 WMMAs).
//  - Attention transposed (S^T = K Q^T, O^T = V^T P^T): every fragment is a
//    contiguous global b128 load or a half-wave register shuffle; zero LDS.
// ---------------------------------------------------------------------------

typedef __attribute__((ext_vector_type(16))) __bf16 bf16x16;
typedef __attribute__((ext_vector_type(8)))  __bf16 bf16x8;
typedef __attribute__((ext_vector_type(8)))  float  f32x8;

#define B_   8
#define N_   1024
#define C_   768
#define NH_  12
#define D_   64
#define M_   (B_ * N_)      // 8192
#define K3_  (3 * C_)       // 2304
#define G_   (B_ * NH_)     // 96 attention groups (flat-reshape semantics)

__device__ __forceinline__ f32x8 wmma_bf16(bf16x16 a, bf16x16 b, f32x8 c) {
  return __builtin_amdgcn_wmma_f32_16x16x32_bf16(false, a, false, b,
                                                 (short)0, c, false, false);
}

// A fragment (16x32 bf16, row-major source, row stride ld).
// Contiguous k order, same as B: lanes 0-15 (m=lane) hold K = k0+[0..15];
//      lanes 16-31 (m=lane-16) hold K = k0+[16..31].
__device__ __forceinline__ bf16x16 load_a_frag(const __bf16* base, int ld,
                                               int k0, int lane) {
  int m  = lane & 15;
  int kb = k0 + ((lane < 16) ? 0 : 16);
  const __bf16* p = base + (size_t)m * ld + kb;
  union { bf16x16 v; bf16x8 h[2]; } u;
  u.h[0] = *reinterpret_cast<const bf16x8*>(p);
  u.h[1] = *reinterpret_cast<const bf16x8*>(p + 8);
  return u.v;
}

// B fragment for out = A * W^T: B[k][n] = W[n][k], W row-major (row stride ld).
// Contiguous k order, same as A: lanes 0-15 (n=lane) hold K = k0+[0..15]; lanes 16-31 hold k0+[16..31].
__device__ __forceinline__ bf16x16 load_bT_frag(const __bf16* base, int ld,
                                                int k0, int lane) {
  int n  = lane & 15;
  int kb = k0 + ((lane < 16) ? 0 : 16);
  const __bf16* p = base + (size_t)n * ld + kb;
  union { bf16x16 v; bf16x8 h[2]; } u;
  u.h[0] = *reinterpret_cast<const bf16x8*>(p);
  u.h[1] = *reinterpret_cast<const bf16x8*>(p + 8);
  return u.v;
}

// ---------------------------------------------------------------------------
// f32 -> bf16 conversion
// ---------------------------------------------------------------------------
__global__ void mhsa_cvt_bf16(const float* __restrict__ src,
                              __bf16* __restrict__ dst, int n) {
  int i = blockIdx.x * 256 + threadIdx.x;
  if (i < n) dst[i] = (__bf16)src[i];
}

// ---------------------------------------------------------------------------
// 2D rotary tables: cos/sin [N][32].  Hs = 32; x = pos%32, y = pos/32.
// column p: j = p/2, freq = 10000^(-(4j)/64); even p -> x*f, odd p -> y*f.
// ---------------------------------------------------------------------------
__global__ void mhsa_rope_tables(float* __restrict__ cosb,
                                 float* __restrict__ sinb) {
  int pos = blockIdx.x;      // 0..1023
  int p   = threadIdx.x;     // 0..31
  int j   = p >> 1;
  float f     = powf(10000.0f, -(4.0f * (float)j) / 64.0f);
  float coord = (p & 1) ? (float)(pos >> 5) : (float)(pos & 31);
  float a = coord * f;
  cosb[pos * 32 + p] = cosf(a);
  sinb[pos * 32 + p] = sinf(a);
}

// ---------------------------------------------------------------------------
// Fused QKV GEMM + rotary.  One wave = 32 rows x 64 cols of qkv.
// Rotary position follows the reference's flat reshape: pos' = (n*12+h)%1024.
// q is pre-scaled by d^-0.5.  q,k written row-major; V written TRANSPOSED
// per group as Vt[g][dd][key] so attention can A-fragment it contiguously.
// ---------------------------------------------------------------------------
__global__ void mhsa_qkv_rope(const __bf16* __restrict__ xb,
                              const __bf16* __restrict__ wqkvb,
                              const float* __restrict__ cosb,
                              const float* __restrict__ sinb,
                              __bf16* __restrict__ qb,
                              __bf16* __restrict__ kb,
                              __bf16* __restrict__ vtb) {
  int wave  = blockIdx.x;
  int mt    = wave / 36;      // 32-row tile over M=8192 (256 tiles)
  int strip = wave % 36;      // 64-col strip over 3C; s=strip/12, h=strip%12
  int lane  = threadIdx.x;

  const __bf16* arow = xb + (size_t)(mt * 32) * C_;
  f32x8 acc[2][4] = {};
  for (int k0 = 0; k0 < C_; k0 += 32) {
    bf16x16 a0 = load_a_frag(arow, C_, k0, lane);
    bf16x16 a1 = load_a_frag(arow + (size_t)16 * C_, C_, k0, lane);
#pragma unroll
    for (int t = 0; t < 4; t++) {
      const __bf16* wrow = wqkvb + (size_t)(strip * 64 + t * 16) * C_;
      bf16x16 b = load_bT_frag(wrow, C_, k0, lane);
      acc[0][t] = wmma_bf16(a0, b, acc[0][t]);
      acc[1][t] = wmma_bf16(a1, b, acc[1][t]);
    }
  }

  // D layout -> LDS (f32), then epilogue one full row per lane.
  __shared__ float tile[32][65];
  {
    int n     = lane & 15;
    int mbase = (lane >> 4) << 3;
#pragma unroll
    for (int mi = 0; mi < 2; mi++)
#pragma unroll
      for (int t = 0; t < 4; t++)
#pragma unroll
        for (int r = 0; r < 8; r++)
          tile[mi * 16 + mbase + r][t * 16 + n] = acc[mi][t][r];
  }
  __syncthreads();

  int s = strip / 12, h = strip % 12;
  int row = lane;                 // 0..31 within tile
  int gm  = mt * 32 + row;        // global x row = b*1024 + n
  int b   = gm >> 10;
  int n_  = gm & 1023;
  int scrr = n_ * 12 + h;         // scrambled per-batch row (reference reshape)
  int pos  = scrr & 1023;         // rotary position
  int g    = b * 12 + (scrr >> 10);  // global attention group

  if (s < 2) {
    __bf16* dst = ((s == 0) ? qb : kb) + (size_t)gm * C_ + h * 64;
    const float* cr = cosb + pos * 32;
    const float* sr = sinb + pos * 32;
    float scl = (s == 0) ? 0.125f : 1.0f;  // fold d^-0.5 into q
#pragma unroll
    for (int j = 0; j < 32; j++) {
      int dd = 2 * j;
      float tr = tile[row][dd], ti = tile[row][dd + 1];
      float c = cr[j], sn = sr[j];
      dst[dd]     = (__bf16)((tr * c - ti * sn) * scl);
      dst[dd + 1] = (__bf16)((tr * sn + ti * c) * scl);
    }
  } else {
    // V: transposed scatter into Vt[g][dd][pos]
    __bf16* dst = vtb + ((size_t)g * D_) * N_ + pos;
#pragma unroll
    for (int dd = 0; dd < 64; dd++)
      dst[(size_t)dd * N_] = (__bf16)tile[row][dd];
  }
}

// ---------------------------------------------------------------------------
// Flash attention, fully transposed, zero LDS.
// One wave = one (group g, 16-query tile).  Per 32-key block:
//   S^T(2 tiles) = K(A-frag) x Q^T(B-frag)     -> 4 WMMA
//   per-query online softmax: 1 shfl_xor(16) for max, 1 for sum
//   P^T B-frag built from S^T D-frag via 16 shfl_xor(16)
//   O^T(4 tiles) += Vt(A-frag) x P^T           -> 4 WMMA
// ---------------------------------------------------------------------------
__global__ void mhsa_flash_attn(const __bf16* __restrict__ qb,
                                const __bf16* __restrict__ kb,
                                const __bf16* __restrict__ vtb,
                                __bf16* __restrict__ ctx) {
  int wave = blockIdx.x;
  int g    = wave >> 6;     // 0..95
  int qt   = wave & 63;     // 0..63
  int lane = threadIdx.x;
  bool lo  = (lane < 16);

  const __bf16* Q  = qb  + ((size_t)g * N_ + qt * 16) * D_;
  const __bf16* K  = kb  + (size_t)g * N_ * D_;
  const __bf16* Vt = vtb + (size_t)g * D_ * N_;

  // Q^T B-fragments (queries across lanes), loaded once.
  bf16x16 qf0 = load_bT_frag(Q, D_, 0, lane);
  bf16x16 qf1 = load_bT_frag(Q, D_, 32, lane);

  f32x8 o[4] = {};          // O^T tiles: rows dd = 16t + r + 8*(lane>=16)
  float rmax = -1e30f;      // per-query (per-lane) stats
  float rsum = 0.0f;

  for (int kt = 0; kt < N_; kt += 32) {
    // prefetch next key block (speculative; OOB prefetches are dropped)
    __builtin_prefetch(K + (size_t)(kt + 32 + (lane & 15)) * D_, 0, 1);
    __builtin_prefetch(Vt + (size_t)(lane & 15) * N_ + kt + 32, 0, 1);

    // ---- S^T = K * Q^T for 32 keys (two 16x16 tiles, keys = M) ----
    f32x8 s0 = {}, s1 = {};
    {
      bf16x16 a00 = load_a_frag(K + (size_t)kt * D_,        D_, 0,  lane);
      bf16x16 a01 = load_a_frag(K + (size_t)kt * D_,        D_, 32, lane);
      bf16x16 a10 = load_a_frag(K + (size_t)(kt + 16) * D_, D_, 0,  lane);
      bf16x16 a11 = load_a_frag(K + (size_t)(kt + 16) * D_, D_, 32, lane);
      s0 = wmma_bf16(a00, qf0, s0);
      s0 = wmma_bf16(a01, qf1, s0);
      s1 = wmma_bf16(a10, qf0, s1);
      s1 = wmma_bf16(a11, qf1, s1);
    }

    // ---- per-query online softmax (query = lane & 15) ----
    float mx = s0[0];
#pragma unroll
    for (int r = 1; r < 8; r++) mx = fmaxf(mx, s0[r]);
#pragma unroll
    for (int r = 0; r < 8; r++) mx = fmaxf(mx, s1[r]);
    mx = fmaxf(mx, __shfl_xor(mx, 16, 32));
    float nm    = fmaxf(rmax, mx);
    float alpha = __expf(rmax - nm);
    float ls = 0.0f;
#pragma unroll
    for (int r = 0; r < 8; r++) { s0[r] = __expf(s0[r] - nm); ls += s0[r]; }
#pragma unroll
    for (int r = 0; r < 8; r++) { s1[r] = __expf(s1[r] - nm); ls += s1[r]; }
    ls += __shfl_xor(ls, 16, 32);
    rsum = rsum * alpha + ls;
    rmax = nm;
#pragma unroll
    for (int t = 0; t < 4; t++)
#pragma unroll
      for (int r = 0; r < 8; r++) o[t][r] = o[t][r] * alpha;

    // ---- P^T B-fragment from S^T D-fragments: half-wave swap only ----
    // B layout needs lane<16: keys kt+[0..15]; lane>=16: keys kt+[16..31].
    // D layout holds keys (r + 8*half) at column (lane&15).
    bf16x16 pf;
#pragma unroll
    for (int r = 0; r < 8; r++) {
      float x0 = __shfl_xor(s0[r], 16, 32);   // partner's T0 row
      float x1 = __shfl_xor(s1[r], 16, 32);   // partner's T1 row
      pf[r]     = (__bf16)(lo ? s0[r] : x1);
      pf[r + 8] = (__bf16)(lo ? x0    : s1[r]);
    }

    // ---- O^T += Vt * P^T (Vt A-fragments: contiguous rows of Vt) ----
#pragma unroll
    for (int t = 0; t < 4; t++) {
      bf16x16 vf = load_a_frag(Vt + (size_t)(t * 16) * N_, N_, kt, lane);
      o[t] = wmma_bf16(vf, pf, o[t]);
    }
  }

  // ---- normalize, pack 8 contiguous dd per tile, one b128 store each ----
  float inv = 1.0f / rsum;
  int   n   = lane & 15;
  size_t row = (size_t)g * N_ + qt * 16 + n;      // ctx row (flat reshape)
  int   ddh = lo ? 0 : 8;
#pragma unroll
  for (int t = 0; t < 4; t++) {
    bf16x8 pk;
#pragma unroll
    for (int r = 0; r < 8; r++) pk[r] = (__bf16)(o[t][r] * inv);
    *reinterpret_cast<bf16x8*>(ctx + row * D_ + t * 16 + ddh) = pk;
  }
}

// ---------------------------------------------------------------------------
// Output projection: out = ctx(8192x768) * Wp^T + bias, f32 out.
// One wave = 32x64 output tile.
// ---------------------------------------------------------------------------
__global__ void mhsa_proj(const __bf16* __restrict__ ctx,
                          const __bf16* __restrict__ wprojb,
                          const float* __restrict__ bias,
                          float* __restrict__ out) {
  int wave  = blockIdx.x;
  int mt    = wave / 12;    // 0..255
  int strip = wave % 12;    // 0..11
  int lane  = threadIdx.x;

  const __bf16* arow = ctx + (size_t)(mt * 32) * C_;
  f32x8 acc[2][4] = {};
  for (int k0 = 0; k0 < C_; k0 += 32) {
    bf16x16 a0 = load_a_frag(arow, C_, k0, lane);
    bf16x16 a1 = load_a_frag(arow + (size_t)16 * C_, C_, k0, lane);
#pragma unroll
    for (int t = 0; t < 4; t++) {
      const __bf16* wrow = wprojb + (size_t)(strip * 64 + t * 16) * C_;
      bf16x16 b = load_bT_frag(wrow, C_, k0, lane);
      acc[0][t] = wmma_bf16(a0, b, acc[0][t]);
      acc[1][t] = wmma_bf16(a1, b, acc[1][t]);
    }
  }

  int n     = lane & 15;
  int mbase = (lane >> 4) << 3;
#pragma unroll
  for (int t = 0; t < 4; t++) {
    int col  = strip * 64 + t * 16 + n;
    float bv = bias[col];
#pragma unroll
    for (int mi = 0; mi < 2; mi++)
#pragma unroll
      for (int r = 0; r < 8; r++) {
        int m = mt * 32 + mi * 16 + mbase + r;
        out[(size_t)m * C_ + col] = acc[mi][t][r] + bv;
      }
  }
}

// ---------------------------------------------------------------------------
// Launch
// ---------------------------------------------------------------------------
extern "C" void kernel_launch(void* const* d_in, const int* in_sizes, int n_in,
                              void* d_out, int out_size, void* d_ws, size_t ws_size,
                              hipStream_t stream) {
  const float* x      = (const float*)d_in[0];
  const float* w_qkv  = (const float*)d_in[1];
  const float* w_proj = (const float*)d_in[2];
  const float* b_proj = (const float*)d_in[3];
  float* out = (float*)d_out;
  (void)in_sizes; (void)n_in; (void)out_size; (void)ws_size;

  char* w = (char*)d_ws;
  __bf16* xb     = (__bf16*)w; w += (size_t)M_  * C_ * 2;
  __bf16* wqkvb  = (__bf16*)w; w += (size_t)K3_ * C_ * 2;
  __bf16* wprojb = (__bf16*)w; w += (size_t)C_  * C_ * 2;
  float*  cosb   = (float*)w;  w += (size_t)N_ * 32 * 4;
  float*  sinb   = (float*)w;  w += (size_t)N_ * 32 * 4;
  __bf16* qb     = (__bf16*)w; w += (size_t)M_ * C_ * 2;
  __bf16* kb     = (__bf16*)w; w += (size_t)M_ * C_ * 2;
  __bf16* vtb    = (__bf16*)w; w += (size_t)G_ * D_ * N_ * 2;  // V transposed
  __bf16* ctx    = (__bf16*)w; w += (size_t)M_ * C_ * 2;

  const int nx = M_ * C_;        // 6291456
  const int nw = K3_ * C_;       // 1769472
  const int np = C_ * C_;        // 589824
  mhsa_cvt_bf16<<<(nx + 255) / 256, 256, 0, stream>>>(x, xb, nx);
  mhsa_cvt_bf16<<<(nw + 255) / 256, 256, 0, stream>>>(w_qkv, wqkvb, nw);
  mhsa_cvt_bf16<<<(np + 255) / 256, 256, 0, stream>>>(w_proj, wprojb, np);
  mhsa_rope_tables<<<N_, 32, 0, stream>>>(cosb, sinb);

  mhsa_qkv_rope<<<(M_ / 32) * 36, 32, 0, stream>>>(xb, wqkvb, cosb, sinb,
                                                   qb, kb, vtb);
  mhsa_flash_attn<<<G_ * 64, 32, 0, stream>>>(qb, kb, vtb, ctx);
  mhsa_proj<<<(M_ / 32) * 12, 32, 0, stream>>>(ctx, wprojb, b_proj, out);
}
